// Encoder_37185826849198
// MI455X (gfx1250) — compile-verified, matrix operand order checked by reading, not run on hardware
//
#include <hip/hip_runtime.h>

// Fully-fused LSTM-VAE encoder for gfx1250 (MI455X), wave32.
// One block = 16 batch rows, 256 threads = 8 waves. All four LSTMs + head
// fused; matmuls via v_wmma_f32_16x16x32_f16 with f32 accumulation.
//  - Phase A (168 steps) holds its weight B-fragments resident in VGPRs
//    (9 frags/wave = 72 VGPRs) -> no per-step weight ds_loads.
//  - Biases folded into the GEMM via a constant-1 z element where K has
//    padding slack (p1 @ k=72, f1 @ k=136); otherwise hoisted to registers.
//  - Inputs staged to LDS f16 once (coalesced) so the recurrent loop never
//    touches HBM.
//  - Branch-free activations: v_exp_f32 + v_rcp_f32.
//
// WMMA fragment layout assumptions (CDNA5 ISA 7.12.2, 16-bit data):
//   A (16x32, MxK): lane l holds M=l&15; slot s holds K=(s>>3)*16+(l>>4)*8+(s&7).
//   B (32x16, KxN): lane l holds N=l&15; slot s holds K=(s>>3)*16+(l>>4)*8+(s&7).
//   C/D (16x16 f32): lane l holds col N=l&15, rows M=(l>>4)*8+r.

typedef __attribute__((ext_vector_type(16))) _Float16 v16h;
typedef __attribute__((ext_vector_type(8)))  float    v8f;

#define TIDS  256
#define MROWS 16

__device__ __forceinline__ int afrag(int m, int k) {
  int kch = k >> 5, kk = k & 31;
  int group = kk >> 4, rem = kk & 15;
  int hi = rem >> 3, within = rem & 7;
  return ((kch * 32 + (m + hi * 16)) << 4) + (group * 8 + within);
}

#define WMMA(A, B, C) __builtin_amdgcn_wmma_f32_16x16x32_f16(false, A, false, B, (short)0, C, false, false)

// LDS-weight matmul (phases B1/B2): bias values pre-hoisted into registers.
template<int KCH, int NT>
__device__ __forceinline__ void lstm_matmul_lds(const _Float16* __restrict__ Wfrag,
                                                const float (&bv)[NT / 8],
                                                const _Float16* __restrict__ zbuf,
                                                float* __restrict__ gbuf,
                                                int wave, int lane) {
  v16h a[KCH];
#pragma unroll
  for (int k = 0; k < KCH; ++k)
    a[k] = *(const v16h*)(zbuf + (k * 32 + lane) * 16);
  constexpr int TPW = NT / 8;
#pragma unroll
  for (int ti = 0; ti < TPW; ++ti) {
    const int tile = wave * TPW + ti;
    v8f c = {bv[ti], bv[ti], bv[ti], bv[ti], bv[ti], bv[ti], bv[ti], bv[ti]};
#pragma unroll
    for (int k = 0; k < KCH; ++k) {
      v16h b = *(const v16h*)(Wfrag + ((tile * KCH + k) * 32 + lane) * 16);
      c = WMMA(a[k], b, c);
    }
    *(v8f*)(gbuf + (tile * 16 + (lane & 15)) * 16 + (lane >> 4) * 8) = c;
  }
}

// Pre-permute f32 global weights into f16 B-fragment order in LDS.
template<int KCH, int NT, typename F>
__device__ __forceinline__ void build_wfrag(_Float16* Wfrag, int tid, F fetch) {
  const int total = NT * KCH * 512;
  for (int idx = tid; idx < total; idx += TIDS) {
    int slot = idx & 15;
    int lane = (idx >> 4) & 31;
    int rest = idx >> 9;
    int kc   = rest % KCH;
    int tile = rest / KCH;
    int nn = lane & 15, hi = lane >> 4;
    int kk = (slot >> 3) * 16 + hi * 8 + (slot & 7);
    Wfrag[idx] = (_Float16)fetch(kc * 32 + kk, tile * 16 + nn);
  }
}

__device__ __forceinline__ float sigm(float x) {
  return __builtin_amdgcn_rcpf(1.f + __expf(-x));
}
__device__ __forceinline__ float tanh_fast(float x) {
  return __builtin_amdgcn_rcpf(1.f + __expf(-2.f * x)) * 2.f - 1.f;
}
__device__ __forceinline__ float lstm_cell(float ig, float fg, float gg, float og,
                                           float& cst) {
  cst = sigm(fg) * cst + sigm(ig) * tanh_fast(gg);
  return sigm(og) * tanh_fast(cst);
}

__global__ __launch_bounds__(TIDS)
void encoder_fused_kernel(
    const float* __restrict__ cond, const float* __restrict__ fut,
    const float* __restrict__ p1_wih, const float* __restrict__ p1_whh,
    const float* __restrict__ p1_bih, const float* __restrict__ p1_bhh,
    const float* __restrict__ p2_wih, const float* __restrict__ p2_whh,
    const float* __restrict__ p2_bih, const float* __restrict__ p2_bhh,
    const float* __restrict__ f1_wih, const float* __restrict__ f1_whh,
    const float* __restrict__ f1_bih, const float* __restrict__ f1_bhh,
    const float* __restrict__ f2_wih, const float* __restrict__ f2_whh,
    const float* __restrict__ f2_bih, const float* __restrict__ f2_bhh,
    const float* __restrict__ w_comb, const float* __restrict__ b_comb,
    const float* __restrict__ w_mean, const float* __restrict__ b_mean,
    const float* __restrict__ w_lvar, const float* __restrict__ b_lvar,
    float* __restrict__ out)
{
  extern __shared__ char smem[];
  _Float16* hstore = (_Float16*)smem;                     //  81920 B: A: staged x / B: f1 h
  _Float16* wfrag  = (_Float16*)(smem + 81920);           // 163840 B: weight fragments
  float*    gbuf   = (float*)   (smem + 245760);          //  32768 B: gate outputs
  _Float16* zbuf   = (_Float16*)(smem + 278528);          //   6144 B: A-fragments
  float*    biasb  = (float*)   (smem + 284672);          //   2048 B
  float*    feat   = (float*)   (smem + 286720);          //   6144 B: [16][96] f32
  float*    hcomb  = (float*)   (smem + 292864);          //   4096 B: [16][64] f32
  _Float16* futbuf = (_Float16*)(smem + 296960);          //   5120 B: staged future x

  const int tid  = threadIdx.x;
  const int lane = tid & 31;
  const int wave = tid >> 5;
  const int b0   = blockIdx.x * MROWS;

  //==================== Phase A: p1 (K=96,N=256) + p2 (K=96,N=128), 168 steps ===
  _Float16* wfP1 = wfrag;            // 24576 halves
  _Float16* wfP2 = wfrag + 24576;    // 12288 halves
  _Float16* zb1 = zbuf;              // [x(8) | h1(64) | bias-1 @72] -> 3 chunks
  _Float16* zb2 = zbuf + 1536;       // [h1(64) | h2(32)]            -> 3 chunks
  float* gb1 = gbuf;                 // 4096 f32
  float* gb2 = gbuf + 4096;          // 2048 f32

  build_wfrag<3,16>(wfP1, tid, [=](int k, int n)->float {
    if (k < 8)   return p1_wih[n*8 + k];
    if (k < 72)  return p1_whh[n*64 + (k-8)];
    if (k == 72) return p1_bih[n] + p1_bhh[n];   // bias via constant-1 input
    return 0.f;
  });
  build_wfrag<3,8>(wfP2, tid, [=](int k, int n)->float {
    if (k < 64) return p2_wih[n*64 + k];
    return p2_whh[n*32 + (k-64)];
  });
  for (int i = tid; i < 128; i += TIDS) biasb[i] = p2_bih[i] + p2_bhh[i];
  for (int i = tid; i < 3072; i += TIDS) zbuf[i] = (_Float16)0.f;
  if (tid < MROWS) zb1[afrag(tid, 72)] = (_Float16)1.f;  // bias feed
  // Stage the whole condition tile into LDS f16 (coalesced, once).
  for (int i = tid; i < MROWS * 168 * 8; i += TIDS) {
    int m = i / (168 * 8), r = i % (168 * 8);
    hstore[i] = (_Float16)cond[(size_t)(b0 + m) * (168 * 8) + r];
  }

  float c1[4] = {0,0,0,0}, c2[2] = {0,0};
  __syncthreads();

  // Weight fragments resident in VGPRs for the whole recurrent loop.
  v16h w1[2][3], w2[3];
#pragma unroll
  for (int ti = 0; ti < 2; ++ti)
#pragma unroll
    for (int k = 0; k < 3; ++k)
      w1[ti][k] = *(const v16h*)(wfP1 + (((wave * 2 + ti) * 3 + k) * 32 + lane) * 16);
#pragma unroll
  for (int k = 0; k < 3; ++k)
    w2[k] = *(const v16h*)(wfP2 + ((wave * 3 + k) * 32 + lane) * 16);
  const float bv2 = biasb[wave * 16 + (lane & 15)];

  for (int t = 0; t < 168; ++t) {
    if (tid < 128) {                       // stage x_t (LDS -> A-fragment)
      int m = tid >> 3, kk = tid & 7;
      zb1[afrag(m, kk)] = hstore[m * (168 * 8) + t * 8 + kk];
    }
    __syncthreads();
    {                                       // p1 matmul, weights in VGPRs
      v16h a0 = *(const v16h*)(zb1 + (0 * 32 + lane) * 16);
      v16h a1 = *(const v16h*)(zb1 + (1 * 32 + lane) * 16);
      v16h a2 = *(const v16h*)(zb1 + (2 * 32 + lane) * 16);
#pragma unroll
      for (int ti = 0; ti < 2; ++ti) {
        v8f c = {};
        c = WMMA(a0, w1[ti][0], c);
        c = WMMA(a1, w1[ti][1], c);
        c = WMMA(a2, w1[ti][2], c);
        *(v8f*)(gb1 + ((wave * 2 + ti) * 16 + (lane & 15)) * 16 + (lane >> 4) * 8) = c;
      }
    }
    __syncthreads();
#pragma unroll
    for (int it = 0; it < 4; ++it) {       // p1 cell: 16x64 elems
      int e = tid + it * TIDS;
      int m = e >> 6, j = e & 63;
      float h = lstm_cell(gb1[j*16+m], gb1[(64+j)*16+m],
                          gb1[(128+j)*16+m], gb1[(192+j)*16+m], c1[it]);
      _Float16 hh = (_Float16)h;
      zb1[afrag(m, 8 + j)] = hh;           // recurrent input for p1
      zb2[afrag(m, j)]     = hh;           // feed p2
    }
    __syncthreads();
    {                                       // p2 matmul, weights in VGPRs
      v16h a0 = *(const v16h*)(zb2 + (0 * 32 + lane) * 16);
      v16h a1 = *(const v16h*)(zb2 + (1 * 32 + lane) * 16);
      v16h a2 = *(const v16h*)(zb2 + (2 * 32 + lane) * 16);
      v8f c = {bv2, bv2, bv2, bv2, bv2, bv2, bv2, bv2};
      c = WMMA(a0, w2[0], c);
      c = WMMA(a1, w2[1], c);
      c = WMMA(a2, w2[2], c);
      *(v8f*)(gb2 + (wave * 16 + (lane & 15)) * 16 + (lane >> 4) * 8) = c;
    }
    __syncthreads();
#pragma unroll
    for (int it = 0; it < 2; ++it) {       // p2 cell: 16x32 elems
      int e = tid + it * TIDS;
      int m = e >> 5, j = e & 31;
      float h = lstm_cell(gb2[j*16+m], gb2[(32+j)*16+m],
                          gb2[(64+j)*16+m], gb2[(96+j)*16+m], c2[it]);
      zb2[afrag(m, 64 + j)] = (_Float16)h;
      if (t == 167) feat[m*96 + j] = fmaxf(h, 0.f);   // relu(last_past)
    }
    __syncthreads();
  }

  //==================== Phase B1: f1 (K=160,N=512), 20 steps ====================
  build_wfrag<5,32>(wfrag, tid, [=](int k, int n)->float {
    if (k < 8)    return f1_wih[n*8 + k];
    if (k < 136)  return f1_whh[n*128 + (k-8)];
    if (k == 136) return f1_bih[n] + f1_bhh[n];  // bias via constant-1 input
    return 0.f;
  });
  for (int i = tid; i < 2560; i += TIDS) zbuf[i] = (_Float16)0.f;
  if (tid < MROWS) zbuf[afrag(tid, 136)] = (_Float16)1.f;
  for (int i = tid; i < MROWS * 20 * 8; i += TIDS) {   // stage future tile
    int m = i / 160, r = i % 160;
    futbuf[i] = (_Float16)fut[(size_t)(b0 + m) * 160 + r];
  }
  float cf1[8] = {0,0,0,0,0,0,0,0};
  const float zb4[4] = {0.f, 0.f, 0.f, 0.f};
  __syncthreads();

  for (int t = 0; t < 20; ++t) {
    if (tid < 128) {
      int m = tid >> 3, kk = tid & 7;
      zbuf[afrag(m, kk)] = futbuf[m * 160 + t * 8 + kk];
    }
    __syncthreads();
    lstm_matmul_lds<5,32>(wfrag, zb4, zbuf, gbuf, wave, lane);
    __syncthreads();
#pragma unroll
    for (int it = 0; it < 8; ++it) {       // f1 cell: 16x128 elems
      int e = tid + it * TIDS;
      int m = e >> 7, j = e & 127;
      float h = lstm_cell(gbuf[j*16+m], gbuf[(128+j)*16+m],
                          gbuf[(256+j)*16+m], gbuf[(384+j)*16+m], cf1[it]);
      _Float16 hh = (_Float16)h;
      zbuf[afrag(m, 8 + j)] = hh;
      hstore[(m*20 + t)*128 + j] = hh;     // stage for f2
    }
    __syncthreads();
  }

  //==================== Phase B2: f2 (K=192,N=256), 20 steps ====================
  build_wfrag<6,16>(wfrag, tid, [=](int k, int n)->float {
    if (k < 128) return f2_wih[n*128 + k];
    return f2_whh[n*64 + (k-128)];
  });
  for (int i = tid; i < 256; i += TIDS) biasb[i] = f2_bih[i] + f2_bhh[i];
  for (int i = tid; i < 3072; i += TIDS) zbuf[i] = (_Float16)0.f;
  float cf2[4] = {0,0,0,0};
  __syncthreads();
  float bvf2[2];
#pragma unroll
  for (int ti = 0; ti < 2; ++ti)
    bvf2[ti] = biasb[(wave * 2 + ti) * 16 + (lane & 15)];

  for (int t = 0; t < 20; ++t) {
#pragma unroll
    for (int it = 0; it < 8; ++it) {       // load f1 h_t as A-fragments
      int e = tid + it * TIDS;
      int m = e >> 7, j = e & 127;
      zbuf[afrag(m, j)] = hstore[(m*20 + t)*128 + j];
    }
    __syncthreads();
    lstm_matmul_lds<6,16>(wfrag, bvf2, zbuf, gbuf, wave, lane);
    __syncthreads();
#pragma unroll
    for (int it = 0; it < 4; ++it) {       // f2 cell: 16x64 elems
      int e = tid + it * TIDS;
      int m = e >> 6, j = e & 63;
      float h = lstm_cell(gbuf[j*16+m], gbuf[(64+j)*16+m],
                          gbuf[(128+j)*16+m], gbuf[(192+j)*16+m], cf2[it]);
      zbuf[afrag(m, 128 + j)] = (_Float16)h;
      if (t == 19) feat[m*96 + 32 + j] = fmaxf(h, 0.f);  // relu(last_fut)
    }
    __syncthreads();
  }

  //==================== Head: comb MLP + mean/logvar (f32 VALU, tiny) ==========
#pragma unroll
  for (int it = 0; it < 4; ++it) {
    int e = tid + it * TIDS;
    int m = e >> 6, n = e & 63;
    float acc = b_comb[n];
    for (int k = 0; k < 96; ++k) acc += feat[m*96 + k] * w_comb[n*96 + k];
    hcomb[m*64 + n] = fmaxf(acc, 0.f);
  }
  __syncthreads();
  {
    int m = tid >> 4, j = tid & 15;
    float am = b_mean[j], av = b_lvar[j];
    for (int k = 0; k < 64; ++k) {
      float hv = hcomb[m*64 + k];
      am += hv * w_mean[j*64 + k];
      av += hv * w_lvar[j*64 + k];
    }
    out[(size_t)(b0 + m) * 16 + j]         = am;          // mean
    out[65536 + (size_t)(b0 + m) * 16 + j] = av;          // log_var
  }
}

extern "C" void kernel_launch(void* const* d_in, const int* in_sizes, int n_in,
                              void* d_out, int out_size, void* d_ws, size_t ws_size,
                              hipStream_t stream) {
  (void)in_sizes; (void)n_in; (void)out_size; (void)d_ws; (void)ws_size;
  const float* P[24];
  for (int i = 0; i < 24; ++i) P[i] = (const float*)d_in[i];
  dim3 grid(4096 / MROWS), block(TIDS);
  size_t shmem = 302080;  // 295 KB dynamic LDS (<= 320 KB/WGP on CDNA5)
  hipLaunchKernelGGL(encoder_fused_kernel, grid, block, shmem, stream,
      P[0], P[1], P[2], P[3], P[4], P[5], P[6], P[7], P[8], P[9],
      P[10], P[11], P[12], P[13], P[14], P[15], P[16], P[17],
      P[18], P[19], P[20], P[21], P[22], P[23], (float*)d_out);
}
